// SNN_54795192762739
// MI455X (gfx1250) — matrix operand order repaired and checked by reading, not compiled, not run on hardware
//
#include <hip/hip_runtime.h>

// ---------------------------------------------------------------------------
// SNN (LIF, 2-layer) on MI455X / gfx1250, wave32 + WMMA f16 (hi/lo split).
//   B=4096, I=1024, H=2048, O=128, T=20, tau=2, v_th=1, hard reset.
//
// Pipeline:
//   K0 (x3): one-pass f32 -> (f16 hi, f16 lo) split of x, W1, W2.
//   K1: cur1 = x @ W1^T + b1 (3-product f16-split WMMA, pure-load inner
//       loop) FUSED with the layer-1 LIF recurrence: cur1 is time-invariant,
//       so the whole 20-step spike train is a per-element function of cur1;
//       the epilogue emits 20-bit spike masks directly (no cur1 round trip).
//   K2: per batch-tile: cooperative LDS expansion of mask bits into f16
//       A-fragments (shared by all 8 waves), k-outer / t-inner WMMA with 20
//       register accumulators, then sequential LIF2 + spike sum.
// ---------------------------------------------------------------------------

typedef __attribute__((ext_vector_type(16))) _Float16 v16h;
typedef __attribute__((ext_vector_type(8)))  float    v8f;

#define B_SZ   4096
#define I_SZ   1024
#define H_SZ   2048
#define O_SZ   128
#define TSTEPS 20

// ---------------------------------------------------------------------------
// K0: split f32 array into hi/lo f16 pair:  v ~= hi + lo + O(2^-22 * v)
// ---------------------------------------------------------------------------
__global__ __launch_bounds__(256)
void snn_split_f16(const float* __restrict__ src, _Float16* __restrict__ hi,
                   _Float16* __restrict__ lo, int n) {
    const int i = blockIdx.x * 256 + threadIdx.x;
    if (i >= n) return;
    const float v = src[i];
    const _Float16 h = (_Float16)v;
    hi[i] = h;
    lo[i] = (_Float16)(v - (float)h);
}

// 20-step layer-1 LIF on a constant input current -> 20-bit spike mask.
__device__ __forceinline__ unsigned lif1_mask(float c) {
    float v = 0.0f;
    unsigned m = 0u;
    #pragma unroll
    for (int t = 0; t < TSTEPS; ++t) {
        v = (v + c) * 0.5f;                    // v += (c - v)/tau, tau = 2
        if (v > 1.0f) { m |= (1u << t); v = 0.0f; }
    }
    return m;
}

// ---------------------------------------------------------------------------
// K1: masks[B,H] = LIF1_spike_train( x[B,I] @ W1[H,I]^T + b1[H] )
// Wave = 16(M) x 32(N) strip (2 C tiles). K chunks of 32. Inner loop is
// b128 loads of pre-split f16 + 6 WMMAs (3-term split per tile).
// ---------------------------------------------------------------------------
__global__ __launch_bounds__(256)
void snn_gemm1_lif1(const _Float16* __restrict__ xh, const _Float16* __restrict__ xl,
                    const _Float16* __restrict__ w1h, const _Float16* __restrict__ w1l,
                    const float* __restrict__ b1, unsigned* __restrict__ masks) {
    const int tid    = threadIdx.x;
    const int lane   = tid & 31;
    const int wave   = tid >> 5;
    const int gwave  = blockIdx.x * 8 + wave;
    const int NP     = H_SZ / 32;          // 64 N-tile pairs
    const int mt     = gwave / NP;         // 0..255
    const int ntp    = gwave % NP;         // 0..63
    const int row16  = lane & 15;
    const int hiHalf = lane >> 4;

    const _Float16* ah_row  = xh  + (size_t)(mt * 16 + row16) * I_SZ;
    const _Float16* al_row  = xl  + (size_t)(mt * 16 + row16) * I_SZ;
    const _Float16* b0h_row = w1h + (size_t)(ntp * 32 + row16) * I_SZ;
    const _Float16* b0l_row = w1l + (size_t)(ntp * 32 + row16) * I_SZ;
    const _Float16* b1h_row = w1h + (size_t)(ntp * 32 + 16 + row16) * I_SZ;
    const _Float16* b1l_row = w1l + (size_t)(ntp * 32 + 16 + row16) * I_SZ;

    v8f acc0 = {}, acc1 = {};
    for (int kc = 0; kc < I_SZ / 32; ++kc) {
        const int k0 = kc * 32;
        // A fragment: K = {0..15} (+16 for upper half-wave)
        const int ka = k0 + hiHalf * 16;
        union { uint4 q[2]; v16h h; } ah, al;
        ah.q[0] = *(const uint4*)(ah_row + ka);
        ah.q[1] = *(const uint4*)(ah_row + ka + 8);
        al.q[0] = *(const uint4*)(al_row + ka);
        al.q[1] = *(const uint4*)(al_row + ka + 8);
        // B fragments: 16 contiguous K per lane (column-major B)
        const int kb = k0 + hiHalf * 16;
        const v16h b0h = *(const v16h*)(b0h_row + kb);
        const v16h b0l = *(const v16h*)(b0l_row + kb);
        const v16h b1h = *(const v16h*)(b1h_row + kb);
        const v16h b1l = *(const v16h*)(b1l_row + kb);

        acc0 = __builtin_amdgcn_wmma_f32_16x16x32_f16(false, ah.h, false, b0h,
                                                      (short)0, acc0, false, false);
        acc0 = __builtin_amdgcn_wmma_f32_16x16x32_f16(false, ah.h, false, b0l,
                                                      (short)0, acc0, false, false);
        acc0 = __builtin_amdgcn_wmma_f32_16x16x32_f16(false, al.h, false, b0h,
                                                      (short)0, acc0, false, false);
        acc1 = __builtin_amdgcn_wmma_f32_16x16x32_f16(false, ah.h, false, b1h,
                                                      (short)0, acc1, false, false);
        acc1 = __builtin_amdgcn_wmma_f32_16x16x32_f16(false, ah.h, false, b1l,
                                                      (short)0, acc1, false, false);
        acc1 = __builtin_amdgcn_wmma_f32_16x16x32_f16(false, al.h, false, b1h,
                                                      (short)0, acc1, false, false);
    }
    // Fused epilogue: + bias, 20-step LIF1 recurrence, store spike masks.
    // C layout: VGPR i -> M = i + 8*hiHalf, N = lane%16.
    const int col0 = ntp * 32 + row16;
    const int col1 = col0 + 16;
    const float bias0 = b1[col0];
    const float bias1 = b1[col1];
    #pragma unroll
    for (int i = 0; i < 8; ++i) {
        const int row = mt * 16 + i + 8 * hiHalf;
        masks[(size_t)row * H_SZ + col0] = lif1_mask(acc0[i] + bias0);
        masks[(size_t)row * H_SZ + col1] = lif1_mask(acc1[i] + bias1);
    }
}

// ---------------------------------------------------------------------------
// K2: 20-step layer-2. Block = 16-row batch tile; wave = 16-col O tile.
// All 8 waves share the same A operand, so the 256 threads cooperatively
// expand mask bits -> f16 spike fragments into LDS once per chunk (producer
// thread `tid` writes A-fragment dword index `tid`; mapping is the identity),
// then every wave ds_load_b128's its fragments. W2 is pre-split f16.
// ---------------------------------------------------------------------------
__global__ __launch_bounds__(256)
void snn_step2(const unsigned* __restrict__ masks,
               const _Float16* __restrict__ w2h, const _Float16* __restrict__ w2l,
               const float* __restrict__ b2, float* __restrict__ out) {
    __shared__ __align__(32) unsigned lds_spk[TSTEPS * 256];   // 20 KB

    const int tid    = threadIdx.x;
    const int lane   = tid & 31;
    const int otile  = tid >> 5;           // 0..7
    const int btile  = blockIdx.x;         // 0..255
    const int row16  = lane & 15;
    const int hiHalf = lane >> 4;

    // Producer mapping: tid -> (frag lane = tid>>3, dword j = tid&7)
    const int pj    = tid & 7;
    const int pm    = (tid >> 3) & 15;                       // A row (M)
    const int pg    = tid >> 7;                              // half-wave group
    const int pkoff = 2 * pj + pg * 16;
    const unsigned* prow = masks + (size_t)(btile * 16 + pm) * H_SZ;

    const _Float16* whrow = w2h + (size_t)(otile * 16 + row16) * H_SZ;
    const _Float16* wlrow = w2l + (size_t)(otile * 16 + row16) * H_SZ;

    v8f acc[TSTEPS];
    {
        v8f z = {};
        #pragma unroll
        for (int t = 0; t < TSTEPS; ++t) acc[t] = z;
    }

    for (int kc = 0; kc < H_SZ / 32; ++kc) {
        const int k0 = kc * 32;
        // W2 fragments for this chunk (pre-split hi/lo f16)
        const int kb = k0 + hiHalf * 16;
        const v16h b_hi = *(const v16h*)(whrow + kb);
        const v16h b_lo = *(const v16h*)(wlrow + kb);

        // Cooperative expansion: 2 mask words -> one A dword per t.
        // (b0 | b1<<16) * 0x3C00 packs two exact f16 {0,1} values, no carries.
        const unsigned mk0 = prow[k0 + pkoff];
        const unsigned mk1 = prow[k0 + pkoff + 1];
        #pragma unroll
        for (int t = 0; t < TSTEPS; ++t) {
            const unsigned bits = ((mk0 >> t) & 1u) | (((mk1 >> t) & 1u) << 16);
            lds_spk[t * 256 + tid] = bits * 0x3C00u;
        }
        __syncthreads();

        #pragma unroll
        for (int t = 0; t < TSTEPS; ++t) {
            const v16h a = *(const v16h*)&lds_spk[t * 256 + lane * 8];
            acc[t] = __builtin_amdgcn_wmma_f32_16x16x32_f16(false, a, false, b_hi,
                                                            (short)0, acc[t], false, false);
            acc[t] = __builtin_amdgcn_wmma_f32_16x16x32_f16(false, a, false, b_lo,
                                                            (short)0, acc[t], false, false);
        }
        __syncthreads();   // before next chunk overwrites lds_spk
    }

    // LIF2 + spike accumulation (time order matters here)
    const int   col  = otile * 16 + row16;
    const float bias = b2[col];
    v8f v2 = {}, ssum = {};
    #pragma unroll
    for (int t = 0; t < TSTEPS; ++t) {
        #pragma unroll
        for (int i = 0; i < 8; ++i) {
            const float cur = acc[t][i] + bias;
            const float v   = (v2[i] + cur) * 0.5f;
            const float s   = (v > 1.0f) ? 1.0f : 0.0f;
            ssum[i] += s;
            v2[i]    = (v > 1.0f) ? 0.0f : v;
        }
    }
    #pragma unroll
    for (int i = 0; i < 8; ++i) {
        const int row = btile * 16 + i + 8 * hiHalf;
        out[(size_t)row * O_SZ + col] = ssum[i] * (1.0f / TSTEPS);
    }
}

// ---------------------------------------------------------------------------
extern "C" void kernel_launch(void* const* d_in, const int* in_sizes, int n_in,
                              void* d_out, int out_size, void* d_ws, size_t ws_size,
                              hipStream_t stream) {
    const float* x  = (const float*)d_in[0];   // [4096,1024]
    const float* W1 = (const float*)d_in[1];   // [2048,1024]
    const float* b1 = (const float*)d_in[2];   // [2048]
    const float* W2 = (const float*)d_in[3];   // [128,2048]
    const float* b2 = (const float*)d_in[4];   // [128]
    float*       out = (float*)d_out;          // [4096,128]

    // Workspace layout (bytes):
    char* ws = (char*)d_ws;
    unsigned* masks = (unsigned*)ws;                    // 33.5 MB
    _Float16* xh  = (_Float16*)(ws + (34u << 20));      //  8 MB
    _Float16* xl  = (_Float16*)(ws + (43u << 20));      //  8 MB
    _Float16* w1h = (_Float16*)(ws + (52u << 20));      //  4 MB
    _Float16* w1l = (_Float16*)(ws + (57u << 20));      //  4 MB
    _Float16* w2h = (_Float16*)(ws + (62u << 20));      //  0.5 MB
    _Float16* w2l = (_Float16*)(ws + (63u << 20));      //  0.5 MB

    const int n_x  = B_SZ * I_SZ;   // 4,194,304
    const int n_w1 = H_SZ * I_SZ;   // 2,097,152
    const int n_w2 = O_SZ * H_SZ;   //   262,144

    // K0: one-pass hi/lo splits
    snn_split_f16<<<(n_x  + 255) / 256, 256, 0, stream>>>(x,  xh,  xl,  n_x);
    snn_split_f16<<<(n_w1 + 255) / 256, 256, 0, stream>>>(W1, w1h, w1l, n_w1);
    snn_split_f16<<<(n_w2 + 255) / 256, 256, 0, stream>>>(W2, w2h, w2l, n_w2);

    // K1: 256 M-tiles * 64 N-pairs = 16384 waves -> 2048 blocks of 8 waves
    snn_gemm1_lif1<<<2048, 256, 0, stream>>>(xh, xl, w1h, w1l, b1, masks);

    // K2: 256 batch tiles, 8 O-tile waves each
    snn_step2<<<B_SZ / 16, 256, 0, stream>>>(masks, w2h, w2l, b2, out);
}
